// GraphSAGEModel_1623497638641
// MI455X (gfx1250) — matrix operand order repaired and checked by reading, compiled, not run on hardware
//
#include <hip/hip_runtime.h>

#define N_NODES 200000
#define N_EDGES 500000
#define IN_CH   166
#define HID_CH  64
#define OUT_CH  2
#define KPAD    192   // IN_CH padded to multiple of 32 for 16x16x32 WMMA
#define KTILES  (KPAD / 32)          // 6
#define L1_TILES  (N_NODES / 32)     // 6250 node tiles of 32
#define L1_BLOCKS 625                // persistent blocks; 10 tiles each

typedef __attribute__((ext_vector_type(16))) __bf16 v16bf;
typedef __attribute__((ext_vector_type(8)))  float  v8f;

__device__ __forceinline__ __bf16 bf(float f) { return (__bf16)f; }

__device__ __forceinline__ unsigned pack_bf16(float a, float b) {
    union { __bf16 h[2]; unsigned u; } v;
    v.h[0] = (__bf16)a;
    v.h[1] = (__bf16)b;
    return v.u;
}

__device__ __forceinline__ void atomic_add_f32(float* p, float v) {
    __hip_atomic_fetch_add(p, v, __ATOMIC_RELAXED, __HIP_MEMORY_SCOPE_AGENT);
}

// ---------------------------------------------------------------- utilities
__global__ void fill_zero_kernel(float* __restrict__ p, long long n) {
    long long i = (long long)blockIdx.x * blockDim.x + threadIdx.x;
    long long stride = (long long)gridDim.x * blockDim.x;
    for (; i < n; i += stride) p[i] = 0.0f;
}

__global__ void deg_count_kernel(const long long* __restrict__ dst,
                                 float* __restrict__ cnt) {
    int e = blockIdx.x * blockDim.x + threadIdx.x;
    if (e < N_EDGES) atomic_add_f32(&cnt[(size_t)dst[e]], 1.0f);
}

__global__ void deg_invert_kernel(float* __restrict__ cnt) {
    int i = blockIdx.x * blockDim.x + threadIdx.x;
    if (i < N_NODES) cnt[i] = 1.0f / fmaxf(cnt[i], 1.0f);
}

// one thread per (edge, channel); agg is padded to stride KPAD (tail stays 0)
__global__ void scatter_x_kernel(const long long* __restrict__ src,
                                 const long long* __restrict__ dst,
                                 const float* __restrict__ x,
                                 float* __restrict__ agg) {
    long long idx = (long long)blockIdx.x * blockDim.x + threadIdx.x;
    if (idx >= (long long)N_EDGES * IN_CH) return;
    int e = (int)(idx / IN_CH);
    int c = (int)(idx - (long long)e * IN_CH);
    long long s = src[e], d = dst[e];
    atomic_add_f32(&agg[(size_t)d * KPAD + c], x[(size_t)s * IN_CH + c]);
}

__global__ void scatter_h_kernel(const long long* __restrict__ src,
                                 const long long* __restrict__ dst,
                                 const float* __restrict__ h,
                                 float* __restrict__ agg) {
    long long idx = (long long)blockIdx.x * blockDim.x + threadIdx.x;
    if (idx >= (long long)N_EDGES * HID_CH) return;
    int e = (int)(idx >> 6);        // /64
    int c = (int)(idx & 63);
    long long s = src[e], d = dst[e];
    atomic_add_f32(&agg[(size_t)d * HID_CH + c], h[(size_t)s * HID_CH + c]);
}

// ---------------------------------------------------------------- layer 1 (WMMA)
// h = relu( (agg1 * deg_inv) @ W1_l^T + x @ W1_r^T + b1 )
// Persistent blocks: 256 threads = 8 waves tiled 2(M) x 4(N) -> 32 nodes x 64 hid
// per node-tile; each block handles L1_TILES/L1_BLOCKS = 10 node tiles.
// B (weights) pre-swizzled in LDS to fragment order, ZERO-PADDED for K>=166 --
// this zero padding is what lets A be loaded unguarded past k=165.
__global__ __launch_bounds__(256)
void sage_layer1_kernel(const float* __restrict__ x,     // [N][IN_CH]
                        const float* __restrict__ agg,   // [N][KPAD], tail zeros
                        const float* __restrict__ deg_inv,
                        const float* __restrict__ W1l,   // [HID][IN]
                        const float* __restrict__ W1r,   // [HID][IN]
                        const float* __restrict__ b1,    // [HID]
                        float* __restrict__ hout) {      // [N][HID]
    __shared__ v16bf sBl[KTILES * 2 * HID_CH];   // 24 KB
    __shared__ v16bf sBr[KTILES * 2 * HID_CH];   // 24 KB

    const int tid = threadIdx.x;

    // ---- stage weights, 8 bf16 (one ds_store_b128) per chunk ----
    {
        uint4* dl = (uint4*)sBl;
        uint4* dr = (uint4*)sBr;
        const int CHUNKS = (KPAD * HID_CH) / 8;   // 1536 per weight
        for (int c = tid; c < CHUNKS; c += 256) {
            int i8    = (c & 1) * 8;
            int n     = (c >> 1) & 63;
            int lh    = (c >> 7) & 1;
            int ktile = c >> 8;
            int kbase = ktile * 32 + lh * 8 + i8 * 2;
            float fl[8], fr[8];
            #pragma unroll
            for (int j = 0; j < 8; ++j) {
                int k = kbase + j;
                bool ok = (k < IN_CH);
                fl[j] = ok ? W1l[n * IN_CH + k] : 0.0f;
                fr[j] = ok ? W1r[n * IN_CH + k] : 0.0f;
            }
            uint4 pl, pr;
            pl.x = pack_bf16(fl[0], fl[1]); pl.y = pack_bf16(fl[2], fl[3]);
            pl.z = pack_bf16(fl[4], fl[5]); pl.w = pack_bf16(fl[6], fl[7]);
            pr.x = pack_bf16(fr[0], fr[1]); pr.y = pack_bf16(fr[2], fr[3]);
            pr.z = pack_bf16(fr[4], fr[5]); pr.w = pack_bf16(fr[6], fr[7]);
            dl[c] = pl;
            dr[c] = pr;
        }
    }
    __syncthreads();

    const int lane  = tid & 31;
    const int wave  = tid >> 5;
    const int wm    = wave >> 2;              // 0..1  (node sub-tile)
    const int wn    = wave & 3;               // 0..3  (hidden tile)
    const int nbase = wn * 16;                // hidden base of this wave
    const int lm    = lane & 15;
    const int lh    = lane >> 4;              // K-half this lane holds

    // ---- hoist all B fragments into registers (invariant per wave) ----
    v16bf Bl[KTILES], Br[KTILES];
    #pragma unroll
    for (int kt = 0; kt < KTILES; ++kt) {
        int fi = (kt * 2 + lh) * HID_CH + nbase + lm;
        Bl[kt] = sBl[fi];
        Br[kt] = sBr[fi];
    }
    const float bias = b1[nbase + lm];

    // ---- persistent loop over node tiles (uniform bounds: EXEC all ones) ----
    for (int tile = blockIdx.x; tile < L1_TILES; tile += L1_BLOCKS) {
        const int nb = tile * 32 + wm * 16;           // node base of this wave
        // per-lane row bases, biased by this lane's K-half (8*lh floats)
        const float* xbase = x   + (size_t)(nb + lm) * IN_CH + 8 * lh;
        const float* abase = agg + (size_t)(nb + lm) * KPAD  + 8 * lh;
        const float  dscale = deg_inv[nb + lm];

        // prefetch next tile's rows for this lane (uniform branch)
        int tnext = tile + L1_BLOCKS;
        if (tnext < L1_TILES) {
            const float* xn = x   + (size_t)(tnext * 32 + wm * 16 + lm) * IN_CH;
            const float* an = agg + (size_t)(tnext * 32 + wm * 16 + lm) * KPAD;
            __builtin_prefetch(xn, 0, 1);
            __builtin_prefetch(an, 0, 1);
        }

        v8f acc = {};
        #pragma unroll
        for (int kt = 0; kt < KTILES; ++kt) {
            const int k = kt * 32;
            v16bf ax, aa;
            // A fragment (16-bit 16x32 layout), unguarded: K>=IN_CH lanes of B are 0.
            //   elems 0..7  = K = k + 8*lh + j      -> xbase[k + j]
            //   elems 8..15 = K = k + 16 + 8*lh + j -> xbase[k + 16 + j]
            #pragma unroll
            for (int j = 0; j < 8; ++j) {
                ax[j]     = bf(xbase[k + j]);
                ax[8 + j] = bf(xbase[k + 16 + j]);
                aa[j]     = bf(abase[k + j]      * dscale);
                aa[8 + j] = bf(abase[k + 16 + j] * dscale);
            }
            acc = __builtin_amdgcn_wmma_f32_16x16x32_bf16(false, ax, false, Br[kt],
                                                          (short)0, acc, false, false);
            acc = __builtin_amdgcn_wmma_f32_16x16x32_bf16(false, aa, false, Bl[kt],
                                                          (short)0, acc, false, false);
        }

        // C/D layout: lane -> N = lm; VGPR i -> M = i + 8*lh. Fuse bias + ReLU.
        // Single base pointer; constant immediate offsets i*HID_CH.
        float* hp = hout + (size_t)(nb + 8 * lh) * HID_CH + nbase + lm;
        #pragma unroll
        for (int i = 0; i < 8; ++i) {
            hp[i * HID_CH] = fmaxf(acc[i] + bias, 0.0f);
        }
    }
}

// ---------------------------------------------------------------- layer 2 (64 -> 2)
__global__ __launch_bounds__(256)
void sage_layer2_kernel(const float* __restrict__ h,
                        const float* __restrict__ agg2,
                        const float* __restrict__ deg_inv,
                        const float* __restrict__ W2l,   // [2][64]
                        const float* __restrict__ W2r,   // [2][64]
                        const float* __restrict__ b2,    // [2]
                        float* __restrict__ out) {       // [N][2]
    __shared__ float sW[4 * HID_CH];  // [W2l row0 | W2l row1 | W2r row0 | W2r row1]
    int tid = threadIdx.x;
    if (tid < 128) sW[tid] = W2l[tid];
    else           sW[tid] = W2r[tid - 128];
    __syncthreads();

    int n = blockIdx.x * 256 + tid;
    if (n >= N_NODES) return;
    float di = deg_inv[n];
    const float* hr = h    + (size_t)n * HID_CH;
    const float* ar = agg2 + (size_t)n * HID_CH;
    float o0 = b2[0], o1 = b2[1];
    #pragma unroll 8
    for (int k = 0; k < HID_CH; ++k) {
        float hv = hr[k];
        float av = ar[k] * di;
        o0 = fmaf(av, sW[k],      fmaf(hv, sW[128 + k], o0));
        o1 = fmaf(av, sW[64 + k], fmaf(hv, sW[192 + k], o1));
    }
    out[(size_t)n * 2 + 0] = o0;
    out[(size_t)n * 2 + 1] = o1;
}

// ---------------------------------------------------------------- launcher
extern "C" void kernel_launch(void* const* d_in, const int* in_sizes, int n_in,
                              void* d_out, int out_size, void* d_ws, size_t ws_size,
                              hipStream_t stream) {
    const float*     x   = (const float*)d_in[0];
    const long long* ei  = (const long long*)d_in[1];   // int64 [2][N_EDGES]
    const float*     W1l = (const float*)d_in[2];
    const float*     W1r = (const float*)d_in[3];
    const float*     b1  = (const float*)d_in[4];
    const float*     W2l = (const float*)d_in[5];
    const float*     W2r = (const float*)d_in[6];
    const float*     b2  = (const float*)d_in[7];
    const long long* src = ei;
    const long long* dst = ei + N_EDGES;

    // workspace layout (floats): [cnt/deg_inv][agg1 (stride KPAD)][agg2][h]
    float* ws   = (float*)d_ws;
    float* cnt  = ws;                                        // N_NODES
    float* agg1 = cnt  + N_NODES;                            // N_NODES*KPAD
    float* agg2 = agg1 + (size_t)N_NODES * KPAD;             // N_NODES*HID_CH
    float* hbuf = agg2 + (size_t)N_NODES * HID_CH;           // N_NODES*HID_CH

    const long long zcount = (long long)N_NODES
                           + (long long)N_NODES * KPAD
                           + (long long)N_NODES * HID_CH;

    fill_zero_kernel<<<2048, 256, 0, stream>>>(ws, zcount);

    deg_count_kernel<<<(N_EDGES + 255) / 256, 256, 0, stream>>>(dst, cnt);
    deg_invert_kernel<<<(N_NODES + 255) / 256, 256, 0, stream>>>(cnt);

    {
        long long total = (long long)N_EDGES * IN_CH;
        int blocks = (int)((total + 255) / 256);
        scatter_x_kernel<<<blocks, 256, 0, stream>>>(src, dst, x, agg1);
    }

    sage_layer1_kernel<<<L1_BLOCKS, 256, 0, stream>>>(x, agg1, cnt,
                                                      W1l, W1r, b1, hbuf);

    {
        long long total = (long long)N_EDGES * HID_CH;
        int blocks = (int)((total + 255) / 256);
        scatter_h_kernel<<<blocks, 256, 0, stream>>>(src, dst, hbuf, agg2);
    }

    sage_layer2_kernel<<<(N_NODES + 255) / 256, 256, 0, stream>>>(
        hbuf, agg2, cnt, W2l, W2r, b2, (float*)d_out);
}
